// sLSTMblock_22582938043164
// MI455X (gfx1250) — matrix operand order repaired and checked by reading, not compiled, not run on hardware
//
#include <hip/hip_runtime.h>
#include <math.h>

#define BB   4
#define SS   2048
#define PS   2111          // SS + 63 zero pad rows in front of each batch
#define DDIM 512
#define BS   8192
#define KW   64
#define DB   128
#define DFF  682
#define EPSF 1e-5f

typedef __attribute__((ext_vector_type(16))) __bf16 v16bf;
typedef __attribute__((ext_vector_type(8)))  float  v8f;
typedef __attribute__((ext_vector_type(4)))  unsigned int uv4;
typedef __attribute__((ext_vector_type(8)))  int iv8;
typedef __attribute__((ext_vector_type(4)))  int iv4;

union FragU {
    v16bf h;
    unsigned int u[8];
};

__device__ __forceinline__ unsigned short f2bf(float f) {
    union { float f; unsigned u; } v; v.f = f;
    unsigned r = v.u + 0x7FFFu + ((v.u >> 16) & 1u);   // round-to-nearest-even
    return (unsigned short)(r >> 16);
}

__device__ __forceinline__ v8f wmma_bf16(const FragU& a, const FragU& b, v8f c) {
    return __builtin_amdgcn_wmma_f32_16x16x32_bf16(
        false, a.h, false, b.h, (short)0, c, false, false);
}

// TDM: DMA a 2D tile (rows x rowUnits, 8-byte units) from global into LDS.
// D# layout per CDNA5 ISA 8.3/8.4 (group0: count/lds/global/type, group1: dims).
// 6-arg builtin form (clang-23 / therock-10.0 headers).
__device__ __forceinline__ void tdm_load_2d(unsigned lds_off, unsigned long long gaddr,
                                            unsigned rowUnits, unsigned rows,
                                            unsigned strideUnits)
{
    uv4 g0;
    g0[0] = 1u;                                              // count=1, user desc
    g0[1] = lds_off;                                         // LDS byte address
    g0[2] = (unsigned)(gaddr & 0xFFFFFFFFull);               // global_addr lo
    g0[3] = (unsigned)((gaddr >> 32) & 0x01FFFFFFull) | (2u << 30); // hi | type=2
    iv8 g1;
    g1[0] = (int)(3u << 16);                                 // mask=0, data_size=8B
    g1[1] = (int)(rowUnits << 16);                           // tensor_dim0 lo16
    g1[2] = (int)((rows & 0xFFFFu) << 16);                   // tensor_dim1 lo16
    g1[3] = (int)((rowUnits & 0xFFFFu) << 16);               // tile_dim0
    g1[4] = (int)(rows & 0xFFFFu);                           // tile_dim1, tile_dim2=0
    g1[5] = (int)strideUnits;                                // tensor_dim0_stride lo32
    g1[6] = 0;
    g1[7] = 0;
    iv4 z4 = {0, 0, 0, 0};
    iv8 z8 = {0, 0, 0, 0, 0, 0, 0, 0};
    __builtin_amdgcn_tensor_load_to_lds(g0, g1, z4, z4, z8, 0);
}

// ---------------------------------------------------------------- zero sums
__global__ void k_zero(float* p) {
    p[blockIdx.x * 256 + threadIdx.x] = 0.f;
}

// --------------------------------------------- zero the 63 pad rows per batch
__global__ void k_zeropad(unsigned int* xnpad_u32) {
    int pr = blockIdx.x;                 // 0..251
    int b = pr / 63, r = pr % 63;
    xnpad_u32[((size_t)b * PS + r) * 256 + threadIdx.x] = 0u;
}

// ------------------------------------------------- input LN -> padded bf16
__global__ void __launch_bounds__(256) k_ln_in(
    const float* __restrict__ x, const float* __restrict__ w,
    const float* __restrict__ b, unsigned short* __restrict__ xnpad)
{
    int lane = threadIdx.x & 31, wid = threadIdx.x >> 5;
    int row  = blockIdx.x * 8 + wid;
    int bb = row >> 11, s = row & (SS - 1);
    const float* xr = x + (size_t)row * DDIM;
    float vals[16]; float sm = 0.f, s2 = 0.f;
#pragma unroll
    for (int i = 0; i < 16; ++i) {
        float v = xr[lane + 32 * i]; vals[i] = v; sm += v; s2 += v * v;
    }
#pragma unroll
    for (int m = 16; m > 0; m >>= 1) { sm += __shfl_xor(sm, m, 32); s2 += __shfl_xor(s2, m, 32); }
    float mu  = sm * (1.f / DDIM);
    float inv = rsqrtf(s2 * (1.f / DDIM) - mu * mu + EPSF);
    unsigned short* orow = xnpad + ((size_t)bb * PS + 63 + s) * DDIM;
#pragma unroll
    for (int i = 0; i < 16; ++i) {
        int c = lane + 32 * i;
        orow[c] = f2bf((vals[i] - mu) * inv * w[c] + b[c]);
    }
}

// --------------------------------- conv weights -> bf16 B-fragments
// fragment fi = (k*16 + c)*32 + t ; per-fragment 32 lanes x 8 dwords
__global__ void k_prep_convw(const float* __restrict__ cw, unsigned int* __restrict__ wfrag)
{
    unsigned gid = blockIdx.x * 256u + threadIdx.x;
    unsigned fi = gid >> 8, r = gid & 255u;
    unsigned lane = r >> 3, j = r & 7u;
    unsigned t = fi & 31u, c = (fi >> 5) & 15u, k = fi >> 9;
    unsigned dout = t * 16u + (lane & 15u);
    unsigned hlf  = lane >> 4;
    unsigned din  = c * 32u + hlf * 8u + 2u * j + ((j >> 2) << 3);
    float a0 = cw[((size_t)dout * DDIM + din) * KW + k];
    float a1 = cw[((size_t)dout * DDIM + din + 1) * KW + k];
    wfrag[gid] = (unsigned)f2bf(a0) | ((unsigned)f2bf(a1) << 16);
}

// ------------------------------- gate weights -> bf16 B-fragments
__global__ void k_prep_gatew(const float* __restrict__ gW, unsigned int* __restrict__ wgfrag)
{
    unsigned gid = blockIdx.x * 256u + threadIdx.x;
    unsigned fi = gid >> 8, r = gid & 255u;
    unsigned lane = r >> 3, j = r & 7u;
    unsigned t7 = fi & 7u, c = (fi >> 3) & 3u, blk = (fi >> 5) & 3u, g = fi >> 7;
    unsigned dcol = t7 * 16u + (lane & 15u);
    unsigned hlf  = lane >> 4;
    unsigned i0   = c * 32u + hlf * 8u + 2u * j + ((j >> 2) << 3);
    size_t base = (size_t)(g * 4u + blk) * DB;
    float a0 = gW[(base + i0) * DB + dcol];
    float a1 = gW[(base + i0 + 1) * DB + dcol];
    wgfrag[gid] = (unsigned)f2bf(a0) | ((unsigned)f2bf(a1) << 16);
}

// -------------------- rbias[g][d] = gb + block_diag(ht_1, rW[g])
__global__ void k_rbias(const float* __restrict__ rW, const float* __restrict__ gb,
                        const float* __restrict__ ht1, float* __restrict__ rbias)
{
    int gid = blockIdx.x * 256 + threadIdx.x;      // 2048 total
    int g = gid >> 9, d = gid & 511;
    int blk = d >> 7, dc = d & 127;
    const float* w = rW + ((size_t)(g * 4 + blk) * DB) * DB;
    float s = 0.f;
    for (int i = 0; i < DB; ++i) s += ht1[blk * DB + i] * w[i * DB + dc];
    rbias[gid] = s + gb[gid];
}

// ---------------- causal conv: TDM-staged A tile in LDS, WMMA over k, SiLU
__global__ void __launch_bounds__(256) k_conv(
    const unsigned short* __restrict__ xnpad, const unsigned int* __restrict__ wfrag,
    const float* __restrict__ cb, unsigned short* __restrict__ xcbf)
{
    extern __shared__ unsigned short atile[];      // 79 rows x 512 bf16 = 79 KB
    int lane = threadIdx.x & 31, wid = threadIdx.x >> 5;
    int r0 = blockIdx.x * 16;
    int t  = blockIdx.y * 8 + wid;                 // dout tile (16 cols)
    int M  = lane & 15, hlf = lane >> 4;
    int bb = r0 >> 11, s0 = r0 & (SS - 1);

    // DMA the 79-row window (covers s0-63 .. s0+15, zeros come from pad rows)
    if (wid == 0) {
        unsigned lds_off = (unsigned)(size_t)atile;
        unsigned long long ga =
            (unsigned long long)(size_t)(xnpad + ((size_t)bb * PS + s0) * DDIM);
        tdm_load_2d(lds_off, ga, DDIM * 2 / 8, 79, DDIM * 2 / 8);
        __builtin_amdgcn_s_wait_tensorcnt(0);
    }
    __syncthreads();

    v8f acc = {0.f, 0.f, 0.f, 0.f, 0.f, 0.f, 0.f, 0.f};

    for (int k = 0; k < KW; ++k) {
        if (k + 1 < KW)
            __builtin_prefetch(wfrag + ((size_t)((k + 1) * 512) + t) * 256 + lane * 8, 0, 1);
        const unsigned short* arow = atile + (size_t)(k + M) * DDIM;
#pragma unroll 4
        for (int c = 0; c < 16; ++c) {
            FragU a, bf;
            const uint4* p0 = (const uint4*)(arow + c * 32 + hlf * 8);
            uint4 q0 = p0[0], q1 = p0[2];          // K offsets 0..7 and 16..23
            a.u[0] = q0.x; a.u[1] = q0.y; a.u[2] = q0.z; a.u[3] = q0.w;
            a.u[4] = q1.x; a.u[5] = q1.y; a.u[6] = q1.z; a.u[7] = q1.w;
            const uint4* wp = (const uint4*)(wfrag + ((size_t)(k * 16 + c) * 32 + t) * 256 + lane * 8);
            uint4 w0 = wp[0], w1 = wp[1];
            bf.u[0] = w0.x; bf.u[1] = w0.y; bf.u[2] = w0.z; bf.u[3] = w0.w;
            bf.u[4] = w1.x; bf.u[5] = w1.y; bf.u[6] = w1.z; bf.u[7] = w1.w;
            acc = wmma_bf16(a, bf, acc);
        }
    }
    int n = lane & 15;
    int d = t * 16 + n;
    float bias = cb[d];
#pragma unroll
    for (int v = 0; v < 8; ++v) {
        int rr = r0 + (hlf ? 8 : 0) + v;
        float xv = acc[v] + bias;
        float si = xv / (1.f + expf(-xv));          // SiLU
        xcbf[(size_t)rr * DDIM + d] = f2bf(si);
    }
}

// -------- fused gates (WMMA block-diag, TDM-staged inputs) + LN + cell
__global__ void __launch_bounds__(256) k_gates(
    const unsigned short* __restrict__ xnpad, const unsigned short* __restrict__ xcbf,
    const unsigned int* __restrict__ wgfrag, const float* __restrict__ rbias,
    const float* __restrict__ ln_g_w, const float* __restrict__ ln_g_b,
    const float* __restrict__ mt1, const float* __restrict__ ct1, const float* __restrict__ nt1,
    const float* __restrict__ ln_c_w, const float* __restrict__ ln_c_b,
    const float* __restrict__ ln_n_w, const float* __restrict__ ln_n_b,
    float* __restrict__ o_buf, float* __restrict__ sum_ct, float* __restrict__ sum_nt)
{
    extern __shared__ float sm[];
    float* bufA  = sm;                 // 16*512 pre-LN / ct scratch
    float* bufI  = sm + 8192;          // log_i -> i
    float* bufF  = sm + 16384;         // f -> nt
    float* stats = sm + 24576;         // 16 x {mu, inv}
    unsigned short* tileXC = (unsigned short*)(sm + 24608);   // 16x512 bf16
    unsigned short* tileXN = tileXC + 16 * DDIM;              // 16x512 bf16

    int tid = threadIdx.x, lane = tid & 31, wid = tid >> 5;
    int r0 = blockIdx.x * 16;
    int bb = r0 >> 11, s0 = r0 & (SS - 1);
    int M = lane & 15, hlf = lane >> 4;

    if (wid == 0) {
        unsigned long long gxc = (unsigned long long)(size_t)(xcbf + (size_t)r0 * DDIM);
        unsigned long long gxn =
            (unsigned long long)(size_t)(xnpad + ((size_t)bb * PS + 63 + s0) * DDIM);
        tdm_load_2d((unsigned)(size_t)tileXC, gxc, DDIM * 2 / 8, 16, DDIM * 2 / 8);
        tdm_load_2d((unsigned)(size_t)tileXN, gxn, DDIM * 2 / 8, 16, DDIM * 2 / 8);
        __builtin_amdgcn_s_wait_tensorcnt(0);
    }
    __syncthreads();

    for (int g = 0; g < 4; ++g) {
        const unsigned short* tin = (g < 2) ? tileXC : tileXN;
        // ---- block-diagonal WMMA: wave wid covers cols [wid*64, wid*64+64)
        for (int tl = 0; tl < 4; ++tl) {
            int t = wid * 4 + tl;
            int blk = t >> 3;
            v8f acc = {0.f, 0.f, 0.f, 0.f, 0.f, 0.f, 0.f, 0.f};
#pragma unroll
            for (int c = 0; c < 4; ++c) {
                FragU a, bf;
                const uint4* p0 = (const uint4*)(tin + (size_t)M * DDIM + blk * DB + c * 32 + hlf * 8);
                uint4 q0 = p0[0], q1 = p0[2];
                a.u[0] = q0.x; a.u[1] = q0.y; a.u[2] = q0.z; a.u[3] = q0.w;
                a.u[4] = q1.x; a.u[5] = q1.y; a.u[6] = q1.z; a.u[7] = q1.w;
                const uint4* wp = (const uint4*)
                    (wgfrag + ((size_t)((g * 4 + blk) * 4 + c) * 8 + (t & 7)) * 256 + lane * 8);
                uint4 w0 = wp[0], w1 = wp[1];
                bf.u[0] = w0.x; bf.u[1] = w0.y; bf.u[2] = w0.z; bf.u[3] = w0.w;
                bf.u[4] = w1.x; bf.u[5] = w1.y; bf.u[6] = w1.z; bf.u[7] = w1.w;
                acc = wmma_bf16(a, bf, acc);
            }
            int dcol = t * 16 + (lane & 15);
            float rb = rbias[g * DDIM + dcol];
#pragma unroll
            for (int v = 0; v < 8; ++v)
                bufA[((hlf ? 8 : 0) + v) * DDIM + dcol] = acc[v] + rb;
        }
        __syncthreads();
        // ---- per-row LN stats (wave handles 2 rows)
        for (int rr = wid * 2; rr < wid * 2 + 2; ++rr) {
            float s = 0.f, s2 = 0.f;
            for (int i = 0; i < 16; ++i) {
                float v = bufA[rr * DDIM + lane + 32 * i]; s += v; s2 += v * v;
            }
            for (int m = 16; m > 0; m >>= 1) { s += __shfl_xor(s, m, 32); s2 += __shfl_xor(s2, m, 32); }
            if (lane == 0) {
                float mu = s * (1.f / DDIM);
                stats[rr * 2] = mu;
                stats[rr * 2 + 1] = rsqrtf(s2 * (1.f / DDIM) - mu * mu + EPSF);
            }
        }
        __syncthreads();
        // ---- LN + activation elementwise
        const float* lw = ln_g_w + g * DDIM;
        const float* lb = ln_g_b + g * DDIM;
        for (int i = 0; i < 32; ++i) {
            int idx = tid + 256 * i;
            int rr = idx >> 9, d = idx & 511;
            float lnv = (bufA[idx] - stats[rr * 2]) * stats[rr * 2 + 1] * lw[d] + lb[d];
            if (g == 0) {
                bufI[idx] = lnv;                               // log_i
            } else if (g == 1) {
                float li = bufI[idx];
                float lf = lnv + mt1[d];
                float mm = fmaxf(lf, li);
                bufI[idx] = expf(li - mm);                     // i
                bufF[idx] = expf(lf - mm);                     // f
            } else if (g == 2) {
                o_buf[(size_t)(r0 + rr) * DDIM + d] = 1.f / (1.f + expf(-lnv)); // o
            } else {
                float z = tanhf(lnv);
                bufA[idx] = bufF[idx] * ct1[d] + bufI[idx] * z;   // ct
            }
        }
        __syncthreads();
    }

    // ---- LN(ct) column-accumulate (mean over B*S = sum/8192, folded later)
    for (int rr = wid * 2; rr < wid * 2 + 2; ++rr) {
        float s = 0.f, s2 = 0.f;
        for (int i = 0; i < 16; ++i) {
            float v = bufA[rr * DDIM + lane + 32 * i]; s += v; s2 += v * v;
        }
        for (int m = 16; m > 0; m >>= 1) { s += __shfl_xor(s, m, 32); s2 += __shfl_xor(s2, m, 32); }
        if (lane == 0) {
            float mu = s * (1.f / DDIM);
            stats[rr * 2] = mu;
            stats[rr * 2 + 1] = rsqrtf(s2 * (1.f / DDIM) - mu * mu + EPSF);
        }
    }
    __syncthreads();
    for (int cc = tid; cc < DDIM; cc += 256) {
        float a = 0.f;
        for (int rr = 0; rr < 16; ++rr)
            a += (bufA[rr * DDIM + cc] - stats[rr * 2]) * stats[rr * 2 + 1];
        atomicAdd(&sum_ct[cc], a * ln_c_w[cc] + 16.f * ln_c_b[cc]);
    }
    __syncthreads();
    // ---- nt = f*nt1 + i ; then LN(nt) column-accumulate
    for (int i = 0; i < 32; ++i) {
        int idx = tid + 256 * i; int d = idx & 511;
        bufF[idx] = bufF[idx] * nt1[d] + bufI[idx];
    }
    __syncthreads();
    for (int rr = wid * 2; rr < wid * 2 + 2; ++rr) {
        float s = 0.f, s2 = 0.f;
        for (int i = 0; i < 16; ++i) {
            float v = bufF[rr * DDIM + lane + 32 * i]; s += v; s2 += v * v;
        }
        for (int m = 16; m > 0; m >>= 1) { s += __shfl_xor(s, m, 32); s2 += __shfl_xor(s2, m, 32); }
        if (lane == 0) {
            float mu = s * (1.f / DDIM);
            stats[rr * 2] = mu;
            stats[rr * 2 + 1] = rsqrtf(s2 * (1.f / DDIM) - mu * mu + EPSF);
        }
    }
    __syncthreads();
    for (int cc = tid; cc < DDIM; cc += 256) {
        float a = 0.f;
        for (int rr = 0; rr < 16; ++rr)
            a += (bufF[rr * DDIM + cc] - stats[rr * 2]) * stats[rr * 2 + 1];
        atomicAdd(&sum_nt[cc], a * ln_n_w[cc] + 16.f * ln_n_b[cc]);
    }
}

// ------------------- ht = o*(ct/nt), row LN, mean column reduce
__global__ void __launch_bounds__(256) k_ht(
    const float* __restrict__ o_buf, const float* __restrict__ sum_ct,
    const float* __restrict__ sum_nt, const float* __restrict__ ln_h_w,
    const float* __restrict__ ln_h_b, float* __restrict__ sum_ht)
{
    __shared__ float colsum[DDIM];
    int tid = threadIdx.x, lane = tid & 31, wid = tid >> 5;
    colsum[tid] = 0.f; colsum[tid + 256] = 0.f;
    __syncthreads();
    int row = blockIdx.x * 8 + wid;
    const float* orow = o_buf + (size_t)row * DDIM;
    float vals[16]; float s = 0.f, s2 = 0.f;
#pragma unroll
    for (int i = 0; i < 16; ++i) {
        int c = lane + 32 * i;
        float h = orow[c] * (sum_ct[c] / sum_nt[c]);   // mean factors cancel
        vals[i] = h; s += h; s2 += h * h;
    }
#pragma unroll
    for (int m = 16; m > 0; m >>= 1) { s += __shfl_xor(s, m, 32); s2 += __shfl_xor(s2, m, 32); }
    float mu  = s * (1.f / DDIM);
    float inv = rsqrtf(s2 * (1.f / DDIM) - mu * mu + EPSF);
#pragma unroll
    for (int i = 0; i < 16; ++i) {
        int c = lane + 32 * i;
        atomicAdd(&colsum[c], (vals[i] - mu) * inv * ln_h_w[c] + ln_h_b[c]);
    }
    __syncthreads();
    atomicAdd(&sum_ht[tid], colsum[tid]);
    atomicAdd(&sum_ht[tid + 256], colsum[tid + 256]);
}

// ------------------------------- GN + GLU head + projection
__global__ void __launch_bounds__(512) k_tail(
    const float* __restrict__ sum_ht, const float* __restrict__ GN_w, const float* __restrict__ GN_b,
    const float* __restrict__ left_w, const float* __restrict__ left_b,
    const float* __restrict__ right_w, const float* __restrict__ right_b,
    const float* __restrict__ ln_out_w, const float* __restrict__ ln_out_b,
    const float* __restrict__ proj_w, const float* __restrict__ proj_b,
    float* __restrict__ out)
{
    __shared__ float sbuf[DDIM];
    __shared__ float vbuf[DFF];
    __shared__ float redA[16], redB[16];
    int tid = threadIdx.x, lane = tid & 31, wid = tid >> 5;

    float ht = sum_ht[tid] * (1.f / 8192.f);
    float s = ht, s2 = ht * ht;
#pragma unroll
    for (int m = 16; m > 0; m >>= 1) { s += __shfl_xor(s, m, 32); s2 += __shfl_xor(s2, m, 32); }
    if (lane == 0) { redA[wid] = s; redB[wid] = s2; }
    __syncthreads();
    if (tid == 0) {
        float a = 0.f, b = 0.f;
        for (int i = 0; i < 16; ++i) { a += redA[i]; b += redB[i]; }
        redA[0] = a; redB[0] = b;
    }
    __syncthreads();
    float mu  = redA[0] * (1.f / DDIM);
    float inv = rsqrtf(redB[0] * (1.f / DDIM) - mu * mu + EPSF);
    sbuf[tid] = (ht - mu) * inv * GN_w[tid] + GN_b[tid];
    __syncthreads();

    for (int j = tid; j < DFF; j += 512) {
        float dl = 0.f, dr = 0.f;
        for (int k = 0; k < DDIM; ++k) {
            float sv = sbuf[k];
            dl += sv * left_w[(size_t)k * DFF + j];
            dr += sv * right_w[(size_t)k * DFF + j];
        }
        float lv = dl + left_b[j];
        float rv = dr + right_b[j];
        float ge = 0.5f * rv * (1.f + erff(rv * 0.70710678f));  // exact GELU
        vbuf[j] = lv * ge;
    }
    __syncthreads();

    float p = 0.f, p2 = 0.f;
    for (int j = tid; j < DFF; j += 512) { float v = vbuf[j]; p += v; p2 += v * v; }
#pragma unroll
    for (int m = 16; m > 0; m >>= 1) { p += __shfl_xor(p, m, 32); p2 += __shfl_xor(p2, m, 32); }
    if (lane == 0) { redA[wid] = p; redB[wid] = p2; }
    __syncthreads();
    if (tid == 0) {
        float a = 0.f, b = 0.f;
        for (int i = 0; i < 16; ++i) { a += redA[i]; b += redB[i]; }
        redA[0] = a; redB[0] = b;
    }
    __syncthreads();
    float mu2  = redA[0] * (1.f / DFF);
    float inv2 = rsqrtf(redB[0] * (1.f / DFF) - mu2 * mu2 + EPSF);
    for (int j = tid; j < DFF; j += 512)
        vbuf[j] = (vbuf[j] - mu2) * inv2 * ln_out_w[j] + ln_out_b[j];
    __syncthreads();

    float acc = 0.f;
    for (int j = 0; j < DFF; ++j) acc += vbuf[j] * proj_w[(size_t)j * DDIM + tid];
    out[tid] = acc + proj_b[tid];
}

// ----------------------------------------------------------------- launch
extern "C" void kernel_launch(void* const* d_in, const int* in_sizes, int n_in,
                              void* d_out, int out_size, void* d_ws, size_t ws_size,
                              hipStream_t stream)
{
    const float* x        = (const float*)d_in[0];
    const float* nt1      = (const float*)d_in[1];
    const float* ct1      = (const float*)d_in[2];
    const float* ht1      = (const float*)d_in[3];
    const float* mt1      = (const float*)d_in[4];
    const float* ln_w     = (const float*)d_in[5];
    const float* ln_b     = (const float*)d_in[6];
    const float* conv_w   = (const float*)d_in[7];
    const float* conv_b   = (const float*)d_in[8];
    const float* gW       = (const float*)d_in[9];
    const float* gb       = (const float*)d_in[10];
    const float* rW       = (const float*)d_in[11];
    const float* ln_g_w   = (const float*)d_in[12];
    const float* ln_g_b   = (const float*)d_in[13];
    const float* GN_w     = (const float*)d_in[14];
    const float* GN_b     = (const float*)d_in[15];
    const float* ln_c_w   = (const float*)d_in[16];
    const float* ln_c_b   = (const float*)d_in[17];
    const float* ln_n_w   = (const float*)d_in[18];
    const float* ln_n_b   = (const float*)d_in[19];
    const float* ln_h_w   = (const float*)d_in[20];
    const float* ln_h_b   = (const float*)d_in[21];
    const float* left_w   = (const float*)d_in[22];
    const float* left_b   = (const float*)d_in[23];
    const float* right_w  = (const float*)d_in[24];
    const float* right_b  = (const float*)d_in[25];
    const float* ln_out_w = (const float*)d_in[26];
    const float* ln_out_b = (const float*)d_in[27];
    const float* proj_w   = (const float*)d_in[28];
    const float* proj_b   = (const float*)d_in[29];
    float* out = (float*)d_out;

    char* ws = (char*)d_ws;
    size_t off = 0;
    unsigned short* xnpad  = (unsigned short*)(ws + off); off += (size_t)BB * PS * DDIM * 2;
    off = (off + 255) & ~(size_t)255;
    unsigned short* xcbf   = (unsigned short*)(ws + off); off += (size_t)BS * DDIM * 2;
    unsigned int*   wfrag  = (unsigned int*)(ws + off);   off += (size_t)DDIM * DDIM * KW * 2;
    unsigned int*   wgfrag = (unsigned int*)(ws + off);   off += (size_t)4 * 4 * DB * DB * 2;
    float*          rbias  = (float*)(ws + off);          off += (size_t)4 * DDIM * 4;
    float*          o_buf  = (float*)(ws + off);          off += (size_t)BS * DDIM * 4;
    float*          sums   = (float*)(ws + off);          off += (size_t)3 * DDIM * 4;
    float* sum_ct = sums;
    float* sum_nt = sums + DDIM;
    float* sum_ht = sums + 2 * DDIM;

    k_zero<<<6, 256, 0, stream>>>(sums);
    k_zeropad<<<BB * 63, 256, 0, stream>>>((unsigned int*)xnpad);
    k_ln_in<<<BS / 8, 256, 0, stream>>>(x, ln_w, ln_b, xnpad);
    k_prep_convw<<<(DDIM * DDIM * KW / 2) / 256, 256, 0, stream>>>(conv_w, wfrag);
    k_prep_gatew<<<(4 * 4 * DB * DB / 2) / 256, 256, 0, stream>>>(gW, wgfrag);
    k_rbias<<<8, 256, 0, stream>>>(rW, gb, ht1, rbias);
    size_t smem_conv = (size_t)79 * DDIM * 2;                     // 79 KB A tile
    k_conv<<<dim3(BS / 16, DDIM / 128), 256, smem_conv, stream>>>(xnpad, wfrag, conv_b, xcbf);
    size_t smem_gates = (size_t)(3 * 16 * DDIM + 32) * 4 + (size_t)2 * 16 * DDIM * 2;
    k_gates<<<BS / 16, 256, smem_gates, stream>>>(xnpad, xcbf, wgfrag, rbias, ln_g_w, ln_g_b,
                                                  mt1, ct1, nt1, ln_c_w, ln_c_b, ln_n_w, ln_n_b,
                                                  o_buf, sum_ct, sum_nt);
    k_ht<<<BS / 8, 256, 0, stream>>>(o_buf, sum_ct, sum_nt, ln_h_w, ln_h_b, sum_ht);
    k_tail<<<1, 512, 0, stream>>>(sum_ht, GN_w, GN_b, left_w, left_b, right_w, right_b,
                                  ln_out_w, ln_out_b, proj_w, proj_b, out);
    (void)in_sizes; (void)n_in; (void)out_size; (void)ws_size;
}
